// CausalTemporalAttention_44985487458309
// MI455X (gfx1250) — matrix operand order repaired and checked by reading, not compiled, not run on hardware
//
#include <hip/hip_runtime.h>
#include <hip/hip_bf16.h>

// CDNA5 / gfx1250. wave32. All matrix math via v_wmma_f32_16x16x32_bf16.
typedef __bf16 bf16_t;
typedef __attribute__((ext_vector_type(16))) __bf16 bf16x16;
typedef __attribute__((ext_vector_type(8)))  __bf16 bf16x8;
typedef __attribute__((ext_vector_type(8)))  float  f32x8;

static __device__ __forceinline__ f32x8 wmma_bf16(bf16x16 a, bf16x16 b, f32x8 c) {
    // (neg_a, A, neg_b, B, c_mod, C, reuse_a, reuse_b)
    return __builtin_amdgcn_wmma_f32_16x16x32_bf16(false, a, false, b, (short)0, c, false, false);
}

static __device__ __forceinline__ f32x8 zero8() {
    f32x8 z = {0.f,0.f,0.f,0.f,0.f,0.f,0.f,0.f};
    return z;
}

// A-fragment (16x32, M = lane&15): element j holds K = (j&7) + 16*(j>>3) + 8*half
// => two contiguous 8-element runs at k0+8*half and k0+16+8*half.
static __device__ __forceinline__ bf16x16 frag_a_bf16(const bf16_t* row, int k0, int half) {
    bf16x8 lo = *reinterpret_cast<const bf16x8*>(row + k0 + 8*half);
    bf16x8 hi = *reinterpret_cast<const bf16x8*>(row + k0 + 16 + 8*half);
    return __builtin_shufflevector(lo, hi, 0,1,2,3,4,5,6,7,8,9,10,11,12,13,14,15);
}

// B-fragment (32x16, N = lane&15): element j holds K = (j&7) + 16*(j>>3) + 8*half
// => two contiguous 8-element runs at k0+8*half and k0+16+8*half.
static __device__ __forceinline__ bf16x16 frag_b_bf16(const bf16_t* row, int k0, int half) {
    bf16x8 lo = *reinterpret_cast<const bf16x8*>(row + k0 + 8*half);
    bf16x8 hi = *reinterpret_cast<const bf16x8*>(row + k0 + 16 + 8*half);
    return __builtin_shufflevector(lo, hi, 0,1,2,3,4,5,6,7,8,9,10,11,12,13,14,15);
}

// ---------------------------------------------------------------------------
// Kernel 0: one-pass fp32 -> bf16 conversion (x, w_qkv, w_out), 8 elems/thread.
// ---------------------------------------------------------------------------
__global__ __launch_bounds__(256) void cvt_f32_to_bf16_kernel(
    const float* __restrict__ in, bf16_t* __restrict__ out, int n8)
{
    const int i = blockIdx.x * 256 + threadIdx.x;
    if (i >= n8) return;
    float t[8];
    const float4* p = reinterpret_cast<const float4*>(in + (size_t)i * 8);
    *reinterpret_cast<float4*>(t + 0) = p[0];
    *reinterpret_cast<float4*>(t + 4) = p[1];
    bf16x8 r;
#pragma unroll
    for (int j = 0; j < 8; ++j) r[j] = (bf16_t)t[j];
    *reinterpret_cast<bf16x8*>(out + (size_t)i * 8) = r;
}

// ---------------------------------------------------------------------------
// Kernel 1: qkv = X[16384,1024] @ Wqkv^T + b  -> split into Q,K (row-major
// [B,L,H,S,128] bf16) and V pre-transposed ([B,L,H,128,S] bf16).
// Per wave: 32x64 output tile (2 M-tiles x 4 N-tiles), K-step 32, all-bf16.
// ---------------------------------------------------------------------------
__global__ __launch_bounds__(256) void qkv_gemm_kernel(
    const bf16_t* __restrict__ xb, const bf16_t* __restrict__ wq,
    const float* __restrict__ b_qkv,
    bf16_t* __restrict__ Q, bf16_t* __restrict__ Kh, bf16_t* __restrict__ Vt)
{
    const int lane = threadIdx.x & 31;
    const int wave = threadIdx.x >> 5;
    const int half = lane >> 4;
    const int ln   = lane & 15;
    const int job  = blockIdx.x * 8 + wave;   // 24576 jobs = 512 * 48
    const int mg   = job / 48;
    const int ng   = job % 48;
    const int m0   = mg * 32;
    const int n0   = ng * 64;

    f32x8 acc[2][4];
#pragma unroll
    for (int mi = 0; mi < 2; ++mi)
#pragma unroll
        for (int i = 0; i < 4; ++i) acc[mi][i] = zero8();

    const bf16_t* arow0 = xb + (size_t)(m0 + ln) * 1024;
    const bf16_t* arow1 = arow0 + (size_t)16 * 1024;

    for (int k0 = 0; k0 < 1024; k0 += 32) {
        bf16x16 a0 = frag_a_bf16(arow0, k0, half);
        bf16x16 a1 = frag_a_bf16(arow1, k0, half);
#pragma unroll
        for (int i = 0; i < 4; ++i) {
            const bf16_t* brow = wq + (size_t)(n0 + 16*i + ln) * 1024;
            bf16x16 bfr = frag_b_bf16(brow, k0, half);
            acc[0][i] = wmma_bf16(a0, bfr, acc[0][i]);
            acc[1][i] = wmma_bf16(a1, bfr, acc[1][i]);
        }
    }

    // D layout: vgpr r -> M = r + 8*half, N = lane&15
#pragma unroll
    for (int i = 0; i < 4; ++i) {
        const int   f    = n0 + 16*i + ln;
        const float bias = b_qkv[f];
        const int which  = f >> 10;      // 0=q, 1=k, 2=v
        const int e      = f & 1023;
        const int h      = e >> 7;
        const int dd     = e & 127;
#pragma unroll
        for (int mi = 0; mi < 2; ++mi) {
#pragma unroll
            for (int r = 0; r < 8; ++r) {
                const int m  = m0 + mi*16 + r + 8*half;
                const int bl = m >> 8;    // b*L + l
                const int s  = m & 255;
                const bf16_t v = (bf16_t)(acc[mi][i][r] + bias);
                const size_t hb = (size_t)(bl*8 + h);
                if (which == 0)      Q [(hb*256 + (size_t)s)*128 + dd] = v;
                else if (which == 1) Kh[(hb*256 + (size_t)s)*128 + dd] = v;
                else                 Vt[(hb*128 + (size_t)dd)*256 + s] = v;
            }
        }
    }
}

// ---------------------------------------------------------------------------
// Kernel 2: fused scores -> mask/decay -> softmax-over-l -> attn @ V.
// One block per (b, h, s-tile of 16); wave w owns layer l = w.
// Softmax axis is the layer axis (8 values) => fully local per (s,t), so we
// tile t by 64 and keep the 8x16x64 fp32 score tile in LDS.
// ---------------------------------------------------------------------------
__global__ __launch_bounds__(256) void attn_fused_kernel(
    const bf16_t* __restrict__ Q, const bf16_t* __restrict__ Kh,
    const bf16_t* __restrict__ Vt, const float* __restrict__ decay,
    bf16_t* __restrict__ Y)
{
    __shared__ float sc[8][16][69];   // 69-float rows: l-stride hits 8 distinct banks

    const int lane = threadIdx.x & 31;
    const int l    = threadIdx.x >> 5;   // wave id == layer id
    const int half = lane >> 4;
    const int ln   = lane & 15;
    const int bh    = blockIdx.x >> 4;
    const int stile = blockIdx.x & 15;
    const int b = bh >> 3, h = bh & 7;
    const int s0 = stile * 16;

    const size_t hb = (size_t)((b*8 + l)*8 + h);
    const bf16_t* Qb = Q  + hb * 256 * 128;
    const bf16_t* Kb = Kh + hb * 256 * 128;
    const bf16_t* Vb = Vt + hb * 128 * 256;

    const float dec    = decay[(b*12 + l)*12 + h];  // decay head-axis broadcasts on B
    const bool  masked = (h > l);                   // mask L-axes land on (l, h)
    const float scale  = dec * 0.08838834764831845f; // decay / sqrt(128)
    const float maskval = -1e9f * dec;

    f32x8 out[8];
#pragma unroll
    for (int i = 0; i < 8; ++i) out[i] = zero8();

    // Q A-fragments for the 16 s-rows (reused across the whole t loop)
    const bf16_t* qrow = Qb + (size_t)(s0 + ln) * 128;
    bf16x16 aq[4];
#pragma unroll
    for (int kk = 0; kk < 4; ++kk) aq[kk] = frag_a_bf16(qrow, 32*kk, half);

    for (int t0 = 0; t0 < 256; t0 += 64) {
        // ---- phase 1: scores for layer l, t in [t0, t0+64) ----
#pragma unroll
        for (int tt = 0; tt < 4; ++tt) {
            f32x8 acc = zero8();
            if (!masked) {
                const bf16_t* krow = Kb + (size_t)(t0 + tt*16 + ln) * 128;
#pragma unroll
                for (int kk = 0; kk < 4; ++kk)
                    acc = wmma_bf16(aq[kk], frag_b_bf16(krow, 32*kk, half), acc);
            }
#pragma unroll
            for (int r = 0; r < 8; ++r) {
                const int srow = r + 8*half;
                const int tcol = tt*16 + ln;
                sc[l][srow][tcol] = masked ? maskval : acc[r] * scale;
            }
        }
        __syncthreads();

        // ---- phase 2: softmax over the 8 layers, per (s,t) ----
        for (int p = threadIdx.x; p < 16*64; p += 256) {
            const int srow = p >> 6, tcol = p & 63;
            float v[8], m = -3.4e38f;
#pragma unroll
            for (int j = 0; j < 8; ++j) { v[j] = sc[j][srow][tcol]; m = fmaxf(m, v[j]); }
            float sum = 0.f;
#pragma unroll
            for (int j = 0; j < 8; ++j) { v[j] = __expf(v[j] - m); sum += v[j]; }
            const float inv = 1.0f / sum;
#pragma unroll
            for (int j = 0; j < 8; ++j) sc[j][srow][tcol] = v[j] * inv;
        }
        __syncthreads();

        // ---- phase 3: out_l += attn_l[16,64] @ V_l[t0:t0+64, :] ----
        bf16x16 af[2];
#pragma unroll
        for (int kt = 0; kt < 2; ++kt) {
            bf16x16 t;
#pragma unroll
            for (int j = 0; j < 16; ++j) {
                const int tl = 32*kt + (j & 7) + 16*(j >> 3) + 8*half;
                t[j] = (bf16_t)sc[l][ln][tl];
            }
            af[kt] = t;
        }
#pragma unroll
        for (int i = 0; i < 8; ++i) {
            const bf16_t* vrow = Vb + (size_t)(16*i + ln) * 256 + t0;  // V^T row = dd
#pragma unroll
            for (int kt = 0; kt < 2; ++kt)
                out[i] = wmma_bf16(af[kt], frag_b_bf16(vrow, 32*kt, half), out[i]);
        }
        // 2 barriers/iter suffice: waves only write their own sc[l] plane in P1/P3.
    }

    // Store to the post-swapaxes flat layout: row = (b*H + h)*L + l
#pragma unroll
    for (int i = 0; i < 8; ++i) {
        const int dd = 16*i + ln;
#pragma unroll
        for (int r = 0; r < 8; ++r) {
            const int s = s0 + r + 8*half;
            const size_t idx = ((((size_t)(b*8 + h)*8 + l)*256 + s)*128 + dd);
            Y[idx] = (bf16_t)out[i][r];
        }
    }
}

// ---------------------------------------------------------------------------
// Kernel 3: out = Y[16384,1024](bf16) @ Wout^T + b_out  (fp32 result)
// ---------------------------------------------------------------------------
__global__ __launch_bounds__(256) void out_proj_kernel(
    const bf16_t* __restrict__ Y, const bf16_t* __restrict__ wo,
    const float* __restrict__ b_out, float* __restrict__ out)
{
    const int lane = threadIdx.x & 31;
    const int wave = threadIdx.x >> 5;
    const int half = lane >> 4;
    const int ln   = lane & 15;
    const int job  = blockIdx.x * 8 + wave;   // 8192 jobs = 512 * 16
    const int mg   = job / 16;
    const int ng   = job % 16;
    const int m0   = mg * 32;
    const int n0   = ng * 64;

    f32x8 acc[2][4];
#pragma unroll
    for (int mi = 0; mi < 2; ++mi)
#pragma unroll
        for (int i = 0; i < 4; ++i) acc[mi][i] = zero8();

    const bf16_t* arow0 = Y + (size_t)(m0 + ln) * 1024;
    const bf16_t* arow1 = arow0 + (size_t)16 * 1024;

    for (int k0 = 0; k0 < 1024; k0 += 32) {
        bf16x16 a0 = frag_a_bf16(arow0, k0, half);
        bf16x16 a1 = frag_a_bf16(arow1, k0, half);
#pragma unroll
        for (int i = 0; i < 4; ++i) {
            const bf16_t* brow = wo + (size_t)(n0 + 16*i + ln) * 1024;
            bf16x16 bfr = frag_b_bf16(brow, k0, half);
            acc[0][i] = wmma_bf16(a0, bfr, acc[0][i]);
            acc[1][i] = wmma_bf16(a1, bfr, acc[1][i]);
        }
    }

#pragma unroll
    for (int i = 0; i < 4; ++i) {
        const int   f    = n0 + 16*i + ln;
        const float bias = b_out[f];
#pragma unroll
        for (int mi = 0; mi < 2; ++mi) {
#pragma unroll
            for (int r = 0; r < 8; ++r) {
                const int m = m0 + mi*16 + r + 8*half;
                out[(size_t)m * 1024 + f] = acc[mi][i][r] + bias;
            }
        }
    }
}

extern "C" void kernel_launch(void* const* d_in, const int* in_sizes, int n_in,
                              void* d_out, int out_size, void* d_ws, size_t ws_size,
                              hipStream_t stream) {
    const float* x      = (const float*)d_in[0];   // [8,8,256,1024]
    const float* w_qkv  = (const float*)d_in[1];   // [3072,1024]
    const float* b_qkv  = (const float*)d_in[2];   // [3072]
    const float* w_out  = (const float*)d_in[3];   // [1024,1024]
    const float* b_out  = (const float*)d_in[4];   // [1024]
    const float* decay  = (const float*)d_in[5];   // [8,12,12]
    float* out = (float*)d_out;

    // Workspace layout (bf16): Q,K,Vt,Y (32 MB each), Xb 32 MB, Wqb 6 MB, Wob 2 MB
    char* ws = (char*)d_ws;
    const size_t SZ = (size_t)16384 * 1024 * sizeof(bf16_t);  // 32 MB
    bf16_t* Q   = (bf16_t*)(ws + 0 * SZ);
    bf16_t* Kh  = (bf16_t*)(ws + 1 * SZ);
    bf16_t* Vt  = (bf16_t*)(ws + 2 * SZ);
    bf16_t* Y   = (bf16_t*)(ws + 3 * SZ);
    bf16_t* Xb  = (bf16_t*)(ws + 4 * SZ);
    bf16_t* Wqb = (bf16_t*)(ws + 5 * SZ);
    bf16_t* Wob = (bf16_t*)(ws + 5 * SZ + (size_t)3072 * 1024 * sizeof(bf16_t));

    // One-pass fp32 -> bf16 materialization (removes cvts from GEMM hot loops).
    cvt_f32_to_bf16_kernel<<<8192, 256, 0, stream>>>(x,     Xb,  16384 * 1024 / 8);
    cvt_f32_to_bf16_kernel<<<1536, 256, 0, stream>>>(w_qkv, Wqb,  3072 * 1024 / 8);
    cvt_f32_to_bf16_kernel<<< 512, 256, 0, stream>>>(w_out, Wob,  1024 * 1024 / 8);

    qkv_gemm_kernel<<<3072, 256, 0, stream>>>(Xb, Wqb, b_qkv, Q, Kh, Vt);
    attn_fused_kernel<<<1024, 256, 0, stream>>>(Q, Kh, Vt, decay, Y);
    out_proj_kernel<<<1024, 256, 0, stream>>>(Y, Wob, b_out, out);
}
